// HCTS_Chimera_v6_61607010894207
// MI455X (gfx1250) — matrix operand order repaired and checked by reading, not compiled, not run on hardware
//
#include <hip/hip_runtime.h>

// ---------------------------------------------------------------------------
// Types / WMMA helpers (gfx1250, wave32)
// ---------------------------------------------------------------------------
typedef _Float16 v16h __attribute__((ext_vector_type(16)));
typedef _Float16 v8h  __attribute__((ext_vector_type(8)));
typedef _Float16 v4h  __attribute__((ext_vector_type(4)));
typedef float    v8f  __attribute__((ext_vector_type(8)));
typedef float    v4f  __attribute__((ext_vector_type(4)));

__device__ __forceinline__ v8f wmma32(v16h a, v16h b, v8f c) {
  // D = A(16x32 f16) * B(32x16 f16) + C(16x16 f32)
  return __builtin_amdgcn_wmma_f32_16x16x32_f16(false, a, false, b, (short)0, c,
                                                false, false);
}

__device__ __forceinline__ v8f zero8() {
  v8f z = {0.f, 0.f, 0.f, 0.f, 0.f, 0.f, 0.f, 0.f};
  return z;
}

__device__ __forceinline__ v4h cvt4(v4f v) {
  v4h h;
  h.x = (_Float16)v.x; h.y = (_Float16)v.y;
  h.z = (_Float16)v.z; h.w = (_Float16)v.w;
  return h;
}

// A-fragment (16x32, f16) from row-major f16 LDS tile, row stride `ld` halves.
// Lane L holds row (row_base + L%16); K = kb..kb+7 and 16+kb..16+kb+7, kb = (L<16)?0:8.
__device__ __forceinline__ v16h load_frag_a(const _Float16* base, int ld,
                                            int row_base, int k0) {
  int lane = threadIdx.x & 31;
  int row  = row_base + (lane & 15);
  int kb   = (lane < 16) ? 0 : 8;
  const _Float16* p = base + row * ld + k0 + kb;
  v8h lo = *(const v8h*)p;
  v8h hi = *(const v8h*)(p + 16);
  return __builtin_shufflevector(lo, hi, 0, 1, 2, 3, 4, 5, 6, 7, 8, 9, 10, 11,
                                 12, 13, 14, 15);
}

// B-fragment (32x16, f16) from K-contiguous LDS tile Bt[n][k], stride `ld` halves.
// Lane L holds col (n_base + L%16); K = kb..kb+7 and 16+kb..16+kb+7, kb = (L<16)?0:8.
__device__ __forceinline__ v16h load_frag_b(const _Float16* base, int ld,
                                            int n_base, int k0) {
  int lane = threadIdx.x & 31;
  int n    = n_base + (lane & 15);
  int kb   = (lane < 16) ? 0 : 8;
  const _Float16* p = base + n * ld + k0 + kb;
  v8h lo = *(const v8h*)p;
  v8h hi = *(const v8h*)(p + 16);
  return __builtin_shufflevector(lo, hi, 0, 1, 2, 3, 4, 5, 6, 7, 8, 9, 10, 11,
                                 12, 13, 14, 15);
}

__device__ __forceinline__ float gelu_f(float x) {
  return 0.5f * x * (1.f + erff(x * 0.70710678118654752f));
}

// ---------------------------------------------------------------------------
// WMMA GEMM: C[M,N] = act(A[M,K] @ W[K,N] + bias) + res
// Block: 256 threads (8 waves). Block tile 128x64; wave tile 16x64.
// REQUIRES: M % 128 == 0, N % 64 == 0, K % 32 == 0 (true for all calls here:
// M=2048, N in {384,1536,32000}, K in {384,1536}).
// ---------------------------------------------------------------------------
#define BM 128
#define BN 64
#define BK 32

__global__ __launch_bounds__(256) void gemm_kernel(
    const float* __restrict__ A, const float* __restrict__ W,
    const float* __restrict__ bias, const float* __restrict__ res,
    float* __restrict__ C, int M, int N, int K, int act) {
  __shared__ _Float16 Ash[BM][BK];  // 8 KB
  __shared__ _Float16 Bsh[BN][BK];  // 4 KB, Bt[n][k]
  int tid = threadIdx.x, wave = tid >> 5, lane = tid & 31;
  int m0 = blockIdx.y * BM, n0 = blockIdx.x * BN;

  v8f acc[4];
#pragma unroll
  for (int c = 0; c < 4; ++c) acc[c] = zero8();

  for (int k0 = 0; k0 < K; k0 += BK) {
    // prefetch next K-tile (global_prefetch_b8)
    if (k0 + BK < K) {
      __builtin_prefetch(A + (size_t)(m0 + (tid & 127)) * K + k0 + BK, 0, 2);
      __builtin_prefetch(W + (size_t)(k0 + BK + (tid & 31)) * N + n0, 0, 2);
    }
    __syncthreads();
    // A tile: 128x32 f32 = 1024 float4, b128 loads, packed b64 LDS stores
    for (int i = tid; i < (BM * BK) / 4; i += 256) {
      int r = i >> 3, c4 = i & 7;
      v4f vv = *(const v4f*)(A + (size_t)(m0 + r) * K + k0 + c4 * 4);
      *(v4h*)&Ash[r][c4 * 4] = cvt4(vv);
    }
    // B tile: 32x64 f32 rows (n-contiguous b128 loads), transposed store
    for (int i = tid; i < (BN * BK) / 4; i += 256) {
      int kk = i >> 4, n4 = i & 15;
      v4f vv = *(const v4f*)(W + (size_t)(k0 + kk) * N + n0 + n4 * 4);
      Bsh[n4 * 4 + 0][kk] = (_Float16)vv.x;
      Bsh[n4 * 4 + 1][kk] = (_Float16)vv.y;
      Bsh[n4 * 4 + 2][kk] = (_Float16)vv.z;
      Bsh[n4 * 4 + 3][kk] = (_Float16)vv.w;
    }
    __syncthreads();
    v16h af = load_frag_a(&Ash[0][0], BK, wave * 16, 0);
#pragma unroll
    for (int c = 0; c < 4; ++c) {
      v16h bf = load_frag_b(&Bsh[0][0], BK, c * 16, 0);
      acc[c]  = wmma32(af, bf, acc[c]);
    }
  }

  int col = lane & 15, rowoff = (lane < 16) ? 0 : 8;
  int mbase = m0 + wave * 16;
#pragma unroll
  for (int c = 0; c < 4; ++c) {
    int n = n0 + c * 16 + col;
    float bv = bias ? bias[n] : 0.f;
#pragma unroll
    for (int r = 0; r < 8; ++r) {
      int m = mbase + r + rowoff;
      float v = acc[c][r] + bv;
      if (act == 1) v = gelu_f(v);
      if (res) v += res[(size_t)m * N + n];
      C[(size_t)m * N + n] = v;
    }
  }
}

// ---------------------------------------------------------------------------
// Hebbian attention: o = relu(beta * Q Kt) [causal] @ V.  NH=4, HD=96, D=384.
// Block: 8 waves x 16 q-rows = 128 q rows for one (b,h).
// ---------------------------------------------------------------------------
__global__ __launch_bounds__(256) void heb_attn_kernel(
    const float* __restrict__ q, const float* __restrict__ k,
    const float* __restrict__ v, const float* __restrict__ beta_p,
    float* __restrict__ o) {
  const int D_ = 384, HDm = 96;
  const int S_ = 1024;
  int b = blockIdx.z, h = blockIdx.y, q0 = blockIdx.x * 128;
  int tid = threadIdx.x, wave = tid >> 5, lane = tid & 31;
  const float beta = beta_p[0];
  const float* qb = q + ((size_t)b * S_) * D_ + h * HDm;
  const float* kb_ = k + ((size_t)b * S_) * D_ + h * HDm;
  const float* vb = v + ((size_t)b * S_) * D_ + h * HDm;
  float* ob = o + ((size_t)b * S_) * D_ + h * HDm;

  __shared__ _Float16 Qs[128][96];    // 24 KB
  __shared__ _Float16 Ks[32][96];     // 6 KB
  __shared__ _Float16 Vt[96][32];     // 6 KB (Vt[hd][key])
  __shared__ _Float16 Ps[8][16][32];  // 8 KB per-wave probs

  for (int i = tid; i < 128 * 24; i += 256) {  // 96 cols = 24 float4
    int r = i / 24, c4 = i % 24;
    v4f vv = *(const v4f*)(qb + (size_t)(q0 + r) * D_ + c4 * 4);
    *(v4h*)&Qs[r][c4 * 4] = cvt4(vv);
  }

  v8f acc[6];
#pragma unroll
  for (int c = 0; c < 6; ++c) acc[c] = zero8();

  int col = lane & 15, rowoff = (lane < 16) ? 0 : 8;
  int qbase = q0 + wave * 16;
  int kend  = q0 + 128;  // uniform trip count across block

  for (int kb = 0; kb < kend; kb += 32) {
    __syncthreads();
    for (int i = tid; i < 32 * 24; i += 256) {
      int r = i / 24, c4 = i % 24;
      v4f kv = *(const v4f*)(kb_ + (size_t)(kb + r) * D_ + c4 * 4);
      v4f vv = *(const v4f*)(vb + (size_t)(kb + r) * D_ + c4 * 4);
      *(v4h*)&Ks[r][c4 * 4] = cvt4(kv);
      Vt[c4 * 4 + 0][r] = (_Float16)vv.x;
      Vt[c4 * 4 + 1][r] = (_Float16)vv.y;
      Vt[c4 * 4 + 2][r] = (_Float16)vv.z;
      Vt[c4 * 4 + 3][r] = (_Float16)vv.w;
    }
    __syncthreads();
    // scores: two 16-key C tiles, contraction over HD=96 (3 k-steps)
#pragma unroll
    for (int sub = 0; sub < 2; ++sub) {
      v8f sc = zero8();
#pragma unroll
      for (int hc = 0; hc < 3; ++hc) {
        v16h af = load_frag_a(&Qs[0][0], 96, wave * 16, hc * 32);
        v16h bf = load_frag_b(&Ks[0][0], 96, sub * 16, hc * 32);
        sc = wmma32(af, bf, sc);
      }
      int key = kb + sub * 16 + col;
#pragma unroll
      for (int r = 0; r < 8; ++r) {
        int qi = qbase + r + rowoff;
        float sv = sc[r] * beta;
        sv = (key <= qi && sv > 0.f) ? sv : 0.f;  // relu + causal
        Ps[wave][r + rowoff][sub * 16 + col] = (_Float16)sv;
      }
    }
    __syncthreads();
    v16h pf = load_frag_a(&Ps[wave][0][0], 32, 0, 0);
#pragma unroll
    for (int c = 0; c < 6; ++c) {
      v16h vf = load_frag_b(&Vt[0][0], 32, c * 16, 0);
      acc[c]  = wmma32(pf, vf, acc[c]);
    }
  }
#pragma unroll
  for (int c = 0; c < 6; ++c)
#pragma unroll
    for (int r = 0; r < 8; ++r)
      ob[(size_t)(qbase + r + rowoff) * D_ + c * 16 + col] = acc[c][r];
}

// ---------------------------------------------------------------------------
// Softmax cross-attention (flash-style). XNH=8, hd=48 (padded to 64), D=384.
// ---------------------------------------------------------------------------
__global__ __launch_bounds__(256) void mha_attn_kernel(
    const float* __restrict__ q, const float* __restrict__ k,
    const float* __restrict__ v, float* __restrict__ o) {
  const int D_ = 384, S_ = 1024, HDm = 48;
  int b = blockIdx.z, h = blockIdx.y, q0 = blockIdx.x * 128;
  int tid = threadIdx.x, wave = tid >> 5, lane = tid & 31;
  const float sm = 0.14433756729740643f;  // 1/sqrt(48)
  const float* qb = q + ((size_t)b * S_) * D_ + h * HDm;
  const float* kb_ = k + ((size_t)b * S_) * D_ + h * HDm;
  const float* vb = v + ((size_t)b * S_) * D_ + h * HDm;
  float* ob = o + ((size_t)b * S_) * D_ + h * HDm;

  __shared__ _Float16 Qs[128][64];    // 16 KB (cols 48..63 zero)
  __shared__ _Float16 Ks[32][64];     // 4 KB
  __shared__ _Float16 Vt[48][32];     // 3 KB
  __shared__ _Float16 Ps[8][16][32];  // 8 KB

  {
    v4h z4 = {(_Float16)0.f, (_Float16)0.f, (_Float16)0.f, (_Float16)0.f};
    for (int i = tid; i < 128 * 16; i += 256) {  // 64 cols = 16 float4
      int r = i >> 4, c4 = i & 15;
      v4h hv = z4;
      if (c4 < 12)
        hv = cvt4(*(const v4f*)(qb + (size_t)(q0 + r) * D_ + c4 * 4));
      *(v4h*)&Qs[r][c4 * 4] = hv;
    }
  }

  v8f acc[3];
#pragma unroll
  for (int c = 0; c < 3; ++c) acc[c] = zero8();
  float mrow[8], lrow[8];
#pragma unroll
  for (int r = 0; r < 8; ++r) { mrow[r] = -3.0e38f; lrow[r] = 0.f; }

  int col = lane & 15, rowoff = (lane < 16) ? 0 : 8;

  for (int kb = 0; kb < S_; kb += 32) {
    __syncthreads();
    {
      v4h z4 = {(_Float16)0.f, (_Float16)0.f, (_Float16)0.f, (_Float16)0.f};
      for (int i = tid; i < 32 * 16; i += 256) {
        int r = i >> 4, c4 = i & 15;
        v4h hv = z4;
        if (c4 < 12)
          hv = cvt4(*(const v4f*)(kb_ + (size_t)(kb + r) * D_ + c4 * 4));
        *(v4h*)&Ks[r][c4 * 4] = hv;
      }
      for (int i = tid; i < 32 * 12; i += 256) {  // V rows: 48 cols = 12 f4
        int r = i / 12, c4 = i % 12;
        v4f vv = *(const v4f*)(vb + (size_t)(kb + r) * D_ + c4 * 4);
        Vt[c4 * 4 + 0][r] = (_Float16)vv.x;
        Vt[c4 * 4 + 1][r] = (_Float16)vv.y;
        Vt[c4 * 4 + 2][r] = (_Float16)vv.z;
        Vt[c4 * 4 + 3][r] = (_Float16)vv.w;
      }
    }
    __syncthreads();
    v8f s0 = zero8(), s1 = zero8();
#pragma unroll
    for (int hc = 0; hc < 2; ++hc) {
      v16h af = load_frag_a(&Qs[0][0], 64, wave * 16, hc * 32);
      v16h b0 = load_frag_b(&Ks[0][0], 64, 0, hc * 32);
      v16h b1 = load_frag_b(&Ks[0][0], 64, 16, hc * 32);
      s0 = wmma32(af, b0, s0);
      s1 = wmma32(af, b1, s1);
    }
#pragma unroll
    for (int r = 0; r < 8; ++r) {
      float a0 = s0[r] * sm, a1 = s1[r] * sm;
      float cm = fmaxf(a0, a1);  // row max across the 16-lane half
      cm = fmaxf(cm, __shfl_xor(cm, 1));
      cm = fmaxf(cm, __shfl_xor(cm, 2));
      cm = fmaxf(cm, __shfl_xor(cm, 4));
      cm = fmaxf(cm, __shfl_xor(cm, 8));
      float mn    = fmaxf(mrow[r], cm);
      float scale = expf(mrow[r] - mn);
      float p0 = expf(a0 - mn), p1 = expf(a1 - mn);
      float rs = p0 + p1;
      rs += __shfl_xor(rs, 1);
      rs += __shfl_xor(rs, 2);
      rs += __shfl_xor(rs, 4);
      rs += __shfl_xor(rs, 8);
      lrow[r] = lrow[r] * scale + rs;
      mrow[r] = mn;
      acc[0][r] *= scale;
      acc[1][r] *= scale;
      acc[2][r] *= scale;
      Ps[wave][r + rowoff][col]      = (_Float16)p0;
      Ps[wave][r + rowoff][16 + col] = (_Float16)p1;
    }
    __syncthreads();
    v16h pf = load_frag_a(&Ps[wave][0][0], 32, 0, 0);
#pragma unroll
    for (int c = 0; c < 3; ++c) {
      v16h vf = load_frag_b(&Vt[0][0], 32, c * 16, 0);
      acc[c]  = wmma32(pf, vf, acc[c]);
    }
  }
  int qbase = q0 + wave * 16;
#pragma unroll
  for (int c = 0; c < 3; ++c)
#pragma unroll
    for (int r = 0; r < 8; ++r)
      ob[(size_t)(qbase + r + rowoff) * D_ + c * 16 + col] =
          acc[c][r] / lrow[r];
}

// ---------------------------------------------------------------------------
// LayerNorm over D=384, one row per 128-thread block.
// ---------------------------------------------------------------------------
__global__ __launch_bounds__(128) void ln_kernel(const float* __restrict__ x,
                                                 const float* __restrict__ g,
                                                 const float* __restrict__ bb,
                                                 float* __restrict__ out) {
  const int D_ = 384;
  int row = blockIdx.x, tid = threadIdx.x;
  const float* xr = x + (size_t)row * D_;
  __shared__ float sb[128];
  float vals[3];
  float s = 0.f;
#pragma unroll
  for (int j = 0; j < 3; ++j) { vals[j] = xr[tid + 128 * j]; s += vals[j]; }
  sb[tid] = s;
  __syncthreads();
  for (int w = 64; w > 0; w >>= 1) {
    if (tid < w) sb[tid] += sb[tid + w];
    __syncthreads();
  }
  float mean = sb[0] * (1.f / 384.f);
  __syncthreads();
  float vv = 0.f;
#pragma unroll
  for (int j = 0; j < 3; ++j) { float d = vals[j] - mean; vv += d * d; }
  sb[tid] = vv;
  __syncthreads();
  for (int w = 64; w > 0; w >>= 1) {
    if (tid < w) sb[tid] += sb[tid + w];
    __syncthreads();
  }
  float rstd = rsqrtf(sb[0] * (1.f / 384.f) + 1e-5f);
#pragma unroll
  for (int j = 0; j < 3; ++j) {
    int d = tid + 128 * j;
    out[(size_t)row * D_ + d] = (vals[j] - mean) * rstd * g[d] + bb[d];
  }
}

// ---------------------------------------------------------------------------
// Small elementwise / reduction kernels
// ---------------------------------------------------------------------------
__global__ void embed_kernel(const int* __restrict__ toks,
                             const float* __restrict__ emb,
                             float* __restrict__ out) {
  const int D_ = 384;
  int i = blockIdx.x * blockDim.x + threadIdx.x;  // < 2*1024*384
  int d = i % D_, bs = i / D_, s = bs % 1024;
  int tok = toks[bs];
  float div = expf(-(float)(d & ~1) * (9.210340371976184f / 384.f));
  float ang = (float)s * div;
  float pe  = (d & 1) ? cosf(ang) : sinf(ang);
  out[i] = emb[(size_t)tok * D_ + d] * 19.595917942265423f + pe;
}

__global__ void seqmlp_kernel(const float* __restrict__ x1,
                              const float* __restrict__ h,
                              const float* __restrict__ ws1,
                              const float* __restrict__ bs1,
                              const float* __restrict__ ws2,
                              const float* __restrict__ bs2,
                              float* __restrict__ out, int n) {
  int i = blockIdx.x * blockDim.x + threadIdx.x;
  if (i >= n) return;
  float hv = h[i], acc = bs2[0];
#pragma unroll
  for (int e = 0; e < 4; ++e) acc += ws2[e] * gelu_f(hv * ws1[e] + bs1[e]);
  out[i] = x1[i] + acc;
}

__global__ void combine_kernel(const float* __restrict__ x,
                               const float* __restrict__ T,
                               const float* __restrict__ H,
                               const float* __restrict__ alpha,
                               float* __restrict__ out, int n) {
  int i = blockIdx.x * blockDim.x + threadIdx.x;
  if (i >= n) return;
  float w = 1.f / (1.f + expf(-alpha[0]));
  out[i] = x[i] + w * T[i] + (1.f - w) * H[i];
}

__global__ __launch_bounds__(256) void diff2_kernel(
    const float* __restrict__ z1, const float* __restrict__ z2,
    float* __restrict__ part) {
  int blk = blockIdx.x;  // 192 = 2 batches x 96 chunks of 4096
  int b = blk / 96, chunk = blk % 96;
  size_t base = (size_t)b * 1024 * 384 + (size_t)chunk * 4096;
  int tid = threadIdx.x;
  float s = 0.f;
  for (int j = tid; j < 4096; j += 256) {
    float d = z2[base + j] - z1[base + j];
    s += d * d;
  }
  __shared__ float sb[256];
  sb[tid] = s;
  __syncthreads();
  for (int w = 128; w > 0; w >>= 1) {
    if (tid < w) sb[tid] += sb[tid + w];
    __syncthreads();
  }
  if (tid == 0) part[blk] = sb[0];
}

__global__ void meanS_kernel(const float* __restrict__ z,
                             float* __restrict__ out) {
  int i = blockIdx.x * blockDim.x + threadIdx.x;
  if (i >= 2 * 384) return;
  int b = i / 384, d = i % 384;
  float s = 0.f;
  for (int t = 0; t < 1024; ++t) s += z[((size_t)b * 1024 + t) * 384 + d];
  out[i] = s * (1.f / 1024.f);
}

__global__ __launch_bounds__(256) void chb_head_kernel(
    const float* __restrict__ means, const float* __restrict__ part,
    const float* __restrict__ a1w, const float* __restrict__ a1b,
    const float* __restrict__ a2w, const float* __restrict__ a2b,
    float* __restrict__ tot) {
  __shared__ float hb[192];
  int t = threadIdx.x;
  if (t < 192) {
    int b = t / 96, j = t % 96;
    float a = a1b[j];
    for (int dd = 0; dd < 384; ++dd) a += means[b * 384 + dd] * a1w[dd * 96 + j];
    hb[t] = gelu_f(a);
  }
  __syncthreads();
  if (t < 2) {
    float lm = a2b[0];
    for (int j = 0; j < 96; ++j) lm += hb[t * 96 + j] * a2w[j];
    float raw = 0.f;
    for (int c = 0; c < 96; ++c) raw += part[t * 96 + c];
    raw *= (1.f / (1024.f * 384.f));
    tot[t] += raw + 1.f / (1.f + expf(-lm));
  }
}

__global__ void zero4_kernel(float* t) {
  if (threadIdx.x < 4) t[threadIdx.x] = 0.f;
}

__global__ void finalize_kernel(const float* __restrict__ tot,
                                float* __restrict__ tail) {
  int t = threadIdx.x;
  if (t < 2) tail[t] = 0.5f * tot[t];  // enc_diss = tot/2
  else if (t < 4) tail[t] = tot[t];    // gen_diss
}

// ---------------------------------------------------------------------------
// Host-side orchestration
// ---------------------------------------------------------------------------
struct Lin { const float *w, *b; };
struct LnP { const float *g, *b; };
struct Heb { Lin wq, wk, wv, wo; LnP norm; const float* beta; };
struct Trm { LnP norm1; Lin l1, l2; LnP norm_s;
             const float *ws1, *bs1, *ws2, *bs2; };
struct Ueb { Trm trm; Heb heb; const float* alpha; LnP norm; };
struct Chb { Heb heb; Lin a1, a2; };
struct MhaP { Lin wq, wk, wv, wo; };
struct Udb { Trm trm; Heb heb; const float* alpha; LnP n1, n2, n3; MhaP mha;
             Lin f1, f2; };

struct Ctx {
  hipStream_t st;
  float *LNB, *Q, *K, *V, *O, *X1, *FF, *Tb, *Hb, *Z1, *PART, *MEANS, *TOT;
};

static void h_gemm(hipStream_t st, const float* A, const float* W,
                   const float* bias, const float* res, float* C, int M, int N,
                   int K, int act) {
  dim3 g((N + BN - 1) / BN, (M + BM - 1) / BM);
  gemm_kernel<<<g, 256, 0, st>>>(A, W, bias, res, C, M, N, K, act);
}

static void h_ln(hipStream_t st, const float* x, LnP p, float* out, int rows) {
  ln_kernel<<<rows, 128, 0, st>>>(x, p.g, p.b, out);
}

static void h_heb(Ctx& c, const float* x, const Heb& hp, float* out) {
  h_ln(c.st, x, hp.norm, c.LNB, 2048);
  h_gemm(c.st, c.LNB, hp.wq.w, hp.wq.b, nullptr, c.Q, 2048, 384, 384, 0);
  h_gemm(c.st, c.LNB, hp.wk.w, hp.wk.b, nullptr, c.K, 2048, 384, 384, 0);
  h_gemm(c.st, c.LNB, hp.wv.w, hp.wv.b, nullptr, c.V, 2048, 384, 384, 0);
  heb_attn_kernel<<<dim3(8, 4, 2), 256, 0, c.st>>>(c.Q, c.K, c.V, hp.beta, c.O);
  h_gemm(c.st, c.O, hp.wo.w, hp.wo.b, x, out, 2048, 384, 384, 0);
}

static void h_trm(Ctx& c, const float* x, const Trm& t, float* out) {
  h_ln(c.st, x, t.norm1, c.LNB, 2048);
  h_gemm(c.st, c.LNB, t.l1.w, t.l1.b, nullptr, c.FF, 2048, 1536, 384, 1);
  h_gemm(c.st, c.FF, t.l2.w, t.l2.b, x, c.X1, 2048, 384, 1536, 0);
  h_ln(c.st, c.X1, t.norm_s, c.LNB, 2048);
  seqmlp_kernel<<<3072, 256, 0, c.st>>>(c.X1, c.LNB, t.ws1, t.bs1, t.ws2,
                                        t.bs2, out, 2048 * 384);
}

static void h_ueb(Ctx& c, float* x, const Ueb& u) {
  h_trm(c, x, u.trm, c.Tb);
  h_heb(c, x, u.heb, c.Hb);
  combine_kernel<<<3072, 256, 0, c.st>>>(x, c.Tb, c.Hb, u.alpha, c.O,
                                         2048 * 384);
  h_ln(c.st, c.O, u.norm, x, 2048);
}

static void h_chb(Ctx& c, float* x, const Chb& p, float* tot_slot) {
  h_heb(c, x, p.heb, c.Z1);  // z1
  h_heb(c, c.Z1, p.heb, x);  // z2 -> x
  diff2_kernel<<<192, 256, 0, c.st>>>(c.Z1, x, c.PART);
  meanS_kernel<<<3, 256, 0, c.st>>>(x, c.MEANS);
  chb_head_kernel<<<1, 256, 0, c.st>>>(c.MEANS, c.PART, p.a1.w, p.a1.b, p.a2.w,
                                       p.a2.b, tot_slot);
}

static void h_mha(Ctx& c, const float* y, const float* mem, const MhaP& m,
                  float* out) {
  h_gemm(c.st, y, m.wq.w, m.wq.b, nullptr, c.Q, 2048, 384, 384, 0);
  h_gemm(c.st, mem, m.wk.w, m.wk.b, nullptr, c.K, 2048, 384, 384, 0);
  h_gemm(c.st, mem, m.wv.w, m.wv.b, nullptr, c.V, 2048, 384, 384, 0);
  mha_attn_kernel<<<dim3(8, 8, 2), 256, 0, c.st>>>(c.Q, c.K, c.V, c.O);
  h_gemm(c.st, c.O, m.wo.w, m.wo.b, y, out, 2048, 384, 384, 0);
}

static void h_udb(Ctx& c, float* y, const float* mem, const Udb& u) {
  h_trm(c, y, u.trm, c.Tb);
  h_heb(c, y, u.heb, c.Hb);
  combine_kernel<<<3072, 256, 0, c.st>>>(y, c.Tb, c.Hb, u.alpha, c.O,
                                         2048 * 384);
  h_ln(c.st, c.O, u.n1, y, 2048);
  h_mha(c, y, mem, u.mha, c.X1);
  h_ln(c.st, c.X1, u.n2, y, 2048);
  h_gemm(c.st, y, u.f1.w, u.f1.b, nullptr, c.FF, 2048, 1536, 384, 1);
  h_gemm(c.st, c.FF, u.f2.w, u.f2.b, y, c.X1, 2048, 384, 1536, 0);
  h_ln(c.st, c.X1, u.n3, y, 2048);
}

extern "C" void kernel_launch(void* const* d_in, const int* in_sizes, int n_in,
                              void* d_out, int out_size, void* d_ws,
                              size_t ws_size, hipStream_t stream) {
  (void)in_sizes; (void)n_in; (void)out_size; (void)ws_size;
  const int T = 2048;                 // B*S
  const size_t TD = (size_t)T * 384;  // token-major activations

  int p = 0;
  const int* src = (const int*)d_in[p++];
  const int* tgt = (const int*)d_in[p++];
  auto nf = [&]() { return (const float*)d_in[p++]; };
  auto getLin = [&]() { Lin l; l.w = nf(); l.b = nf(); return l; };
  auto getLn  = [&]() { LnP l; l.g = nf(); l.b = nf(); return l; };
  auto getHeb = [&]() {
    Heb h; h.wq = getLin(); h.wk = getLin(); h.wv = getLin(); h.wo = getLin();
    h.norm = getLn(); h.beta = nf(); return h;
  };
  auto getTrm = [&]() {
    Trm t; t.norm1 = getLn(); t.l1 = getLin(); t.l2 = getLin();
    t.norm_s = getLn(); t.ws1 = nf(); t.bs1 = nf(); t.ws2 = nf(); t.bs2 = nf();
    return t;
  };
  auto getUeb = [&]() {
    Ueb u; u.trm = getTrm(); u.heb = getHeb(); u.alpha = nf(); u.norm = getLn();
    return u;
  };
  auto getChb = [&]() {
    Chb c; c.heb = getHeb(); c.a1 = getLin(); c.a2 = getLin(); return c;
  };
  auto getMha = [&]() {
    MhaP m; m.wq = getLin(); m.wk = getLin(); m.wv = getLin(); m.wo = getLin();
    return m;
  };
  auto getUdb = [&]() {
    Udb u; u.trm = getTrm(); u.heb = getHeb(); u.alpha = nf();
    u.n1 = getLn(); u.n2 = getLn(); u.n3 = getLn();
    u.mha = getMha(); u.f1 = getLin(); u.f2 = getLin(); return u;
  };

  const float* emb = nf();
  Ueb syn0 = getUeb(), syn1 = getUeb();
  Ueb sem0 = getUeb(), sem1 = getUeb();
  Chb rea0 = getChb(), rea1 = getChb();
  Udb dec0 = getUdb(), dec1 = getUdb(), dec2 = getUdb(), dec3 = getUdb(),
      dec4 = getUdb();
  Chb dre = getChb();
  Lin fco = getLin();

  float* w = (float*)d_ws;
  size_t off = 0;
  auto alloc = [&](size_t n) { float* r = w + off; off += n; return r; };
  float* X = alloc(TD);  // encoder stream -> becomes mem
  float* Y = alloc(TD);  // decoder stream
  Ctx c;
  c.st    = stream;
  c.Tb    = alloc(TD);
  c.Hb    = alloc(TD);
  c.LNB   = alloc(TD);
  c.Q     = alloc(TD);
  c.K     = alloc(TD);
  c.V     = alloc(TD);
  c.O     = alloc(TD);
  c.X1    = alloc(TD);
  c.Z1    = alloc(TD);
  c.FF    = alloc((size_t)T * 1536);
  c.PART  = alloc(256);
  c.MEANS = alloc(1024);
  c.TOT   = alloc(8);

  zero4_kernel<<<1, 32, 0, stream>>>(c.TOT);

  // ---------------- encoder ----------------
  embed_kernel<<<3072, 256, 0, stream>>>(src, emb, X);
  h_ueb(c, X, syn0);
  h_ueb(c, X, syn1);
  h_ueb(c, X, sem0);
  h_ueb(c, X, sem1);
  h_chb(c, X, rea0, c.TOT);
  h_chb(c, X, rea1, c.TOT);
  // X now == mem

  // ---------------- decoder ----------------
  embed_kernel<<<3072, 256, 0, stream>>>(tgt, emb, Y);
  h_udb(c, Y, X, dec0);
  h_udb(c, Y, X, dec1);
  h_udb(c, Y, X, dec2);
  h_udb(c, Y, X, dec3);
  h_udb(c, Y, X, dec4);
  h_chb(c, Y, dre, c.TOT + 2);

  // logits -> d_out
  h_gemm(stream, Y, fco.w, fco.b, nullptr, (float*)d_out, 2048, 32000, 384, 0);
  finalize_kernel<<<1, 32, 0, stream>>>(c.TOT,
                                        (float*)d_out + (size_t)2048 * 32000);
}
